// SelfAttentionModule_65481071401382
// MI455X (gfx1250) — compile-verified, matrix operand order checked by reading, not run on hardware
//
#include <hip/hip_runtime.h>

typedef __attribute__((ext_vector_type(16))) _Float16 v16h;
typedef __attribute__((ext_vector_type(8)))  _Float16 v8h;
typedef __attribute__((ext_vector_type(4)))  _Float16 v4h;
typedef __attribute__((ext_vector_type(8)))  float    v8f;

#define EMBED 1024
#define HEADS 16
#define HDIM  64
#define BATCH 4
#define SEQ   2048
#define MTOT  (BATCH * SEQ)   // 8192

// ---------------------------------------------------------------------------
// Fragment loader: 16x32 f16 tile from a row-major source where the wave-lane
// indexes the M (or N) row and K is contiguous in memory.
// Layout per CDNA5 ISA 7.12.2 (16-bit A 16x32):
//   lanes 0-15 : row = lane,    elems 0..7 = K[0..7],  elems 8..15 = K[16..23]
//   lanes 16-31: row = lane-16, elems 0..7 = K[8..15], elems 8..15 = K[24..31]
// -> two 16-byte loads per lane. B (32x16) uses the symmetric layout, so the
// same loader serves B fragments from a [N, K] row-major (i.e. B^T) source.
// ---------------------------------------------------------------------------
__device__ __forceinline__ v16h load_frag(const _Float16* __restrict__ p, int ld) {
  const int lane = threadIdx.x & 31;
  const _Float16* q = p + (size_t)(lane & 15) * ld + ((lane >> 4) << 3);
  union { v8h h[2]; v16h v; } u;
  u.h[0] = *(const v8h*)(q);
  u.h[1] = *(const v8h*)(q + 16);
  return u.v;
}

__device__ __forceinline__ v8f wmma32(v16h a, v16h b, v8f c) {
  return __builtin_amdgcn_wmma_f32_16x16x32_f16(
      /*neg_a=*/false, a, /*neg_b=*/false, b,
      /*c_mod=*/(short)0, c, /*reuse_a=*/false, /*reuse_b=*/false);
}

// ---------------------------------------------------------------------------
// f32 -> f16 conversion (4 elements / thread)
// ---------------------------------------------------------------------------
__global__ void __launch_bounds__(256)
cvt_f32_f16(const float* __restrict__ src, _Float16* __restrict__ dst, int n) {
  int i = (blockIdx.x * blockDim.x + threadIdx.x) * 4;
  if (i >= n) return;
  float4 f = *(const float4*)(src + i);
  v4h h;
  h[0] = (_Float16)f.x; h[1] = (_Float16)f.y;
  h[2] = (_Float16)f.z; h[3] = (_Float16)f.w;
  *(v4h*)(dst + i) = h;
}

// dst[c][r] = (f16) src[r][c]   (rows x cols source)
__global__ void __launch_bounds__(256)
transpose_cvt(const float* __restrict__ src, _Float16* __restrict__ dst,
              int rows, int cols) {
  int t = blockIdx.x * blockDim.x + threadIdx.x;
  if (t >= rows * cols) return;
  int r = t / cols, c = t % cols;
  dst[(size_t)c * rows + r] = (_Float16)src[t];
}

// ---------------------------------------------------------------------------
// WMMA GEMM: C[M,N] = A[M,K] * B[K,N], with B supplied transposed as Bt[N,K].
// Each wave computes a 64x64 tile (16 wmmas per K=32 step).
// mode 0: store f16 to [B,H,S,D]      (Q, K projections)
// mode 1: store f16 to [B,H,D,S]      (V projection, pre-transposed for P@V)
// mode 2: store f32 to [M,N] + bias   (final output projection)
// ---------------------------------------------------------------------------
__global__ void __launch_bounds__(256)
gemm_wmma(const _Float16* __restrict__ A, const _Float16* __restrict__ Bt,
          _Float16* __restrict__ outh, float* __restrict__ outf,
          const float* __restrict__ bias, int M, int N, int Kd, int mode) {
  const int wave  = blockIdx.x * (blockDim.x >> 5) + (threadIdx.x >> 5);
  const int tilesN = N >> 6;
  const int tm = wave / tilesN;
  const int tn = wave % tilesN;
  if (tm * 64 >= M) return;
  const int lane = threadIdx.x & 31;
  const int half = lane >> 4, nlo = lane & 15;

  const _Float16* Abase = A  + (size_t)tm * 64 * Kd;
  const _Float16* Bbase = Bt + (size_t)tn * 64 * Kd;

  v8f acc[4][4] = {};

  for (int k0 = 0; k0 < Kd; k0 += 32) {
    // speculative prefetch of the next K-panel (global_prefetch_b8)
    if (k0 + 32 < Kd) {
      __builtin_prefetch(Abase + (size_t)(lane & 15) * Kd + k0 + 32, 0, 1);
      __builtin_prefetch(Bbase + (size_t)(lane & 15) * Kd + k0 + 32, 0, 1);
    }
    v16h af[4];
#pragma unroll
    for (int i = 0; i < 4; ++i)
      af[i] = load_frag(Abase + (size_t)(i * 16) * Kd + k0, Kd);
#pragma unroll
    for (int j = 0; j < 4; ++j) {
      v16h bf = load_frag(Bbase + (size_t)(j * 16) * Kd + k0, Kd);
#pragma unroll
      for (int i = 0; i < 4; ++i)
        acc[i][j] = wmma32(af[i], bf, acc[i][j]);
    }
  }

#pragma unroll
  for (int i = 0; i < 4; ++i) {
#pragma unroll
    for (int j = 0; j < 4; ++j) {
#pragma unroll
      for (int r = 0; r < 8; ++r) {
        const int m = tm * 64 + i * 16 + r + half * 8;
        const int n = tn * 64 + j * 16 + nlo;
        const float v = acc[i][j][r];
        if (mode == 2) {
          outf[(size_t)m * N + n] = v + bias[n];
        } else {
          const int b = m >> 11, s = m & (SEQ - 1);
          const int h = n >> 6,  d = n & (HDIM - 1);
          const size_t idx = (mode == 0)
              ? ((size_t)(b * HEADS + h) * SEQ + s) * HDIM + d
              : ((size_t)(b * HEADS + h) * HDIM + d) * SEQ + s;
          outh[idx] = (_Float16)v;
        }
      }
    }
  }
}

// ---------------------------------------------------------------------------
// Flash-attention: one wave per 16-row Q tile of one (batch, head).
// Q,K: [B,H,S,D] f16. Vt: [B,H,D,S] f16. Ctx: [B,S,E] f16.
// Streams 64 KV rows/iter: 8 score WMMAs, one online-softmax pass (cross-lane
// reductions amortized over 64 columns), 8 ctx WMMAs.
// ---------------------------------------------------------------------------
__global__ void __launch_bounds__(256)
attn_wmma(const _Float16* __restrict__ Q, const _Float16* __restrict__ K,
          const _Float16* __restrict__ Vt, _Float16* __restrict__ Ctx) {
  __shared__ _Float16 ldsP[8][16][64];   // per-wave P staging (C-layout -> A-layout)

  const int wave = blockIdx.x * 8 + (threadIdx.x >> 5);
  const int qt = wave & 127;             // q tile within head (S/16 = 128)
  const int bh = wave >> 7;              // b*HEADS + h
  const int lane = threadIdx.x & 31;
  const int half = lane >> 4, nlo = lane & 15;
  const int w = threadIdx.x >> 5;

  const _Float16* Qb = Q  + ((size_t)bh * SEQ + qt * 16) * HDIM;
  const _Float16* Kb = K  + (size_t)bh * SEQ * HDIM;
  const _Float16* Vb = Vt + (size_t)bh * HDIM * SEQ;

  const v16h aq0 = load_frag(Qb + 0,  HDIM);   // Q rows, d 0..31
  const v16h aq1 = load_frag(Qb + 32, HDIM);   // Q rows, d 32..63

  v8f ctx[4] = {};
  float mrow[8], lrow[8];
#pragma unroll
  for (int r = 0; r < 8; ++r) { mrow[r] = -1e30f; lrow[r] = 0.0f; }
  const float sc = 0.125f;               // 1/sqrt(64)

  for (int kb = 0; kb < SEQ; kb += 64) {
    // prefetch next KV panel into cache hierarchy
    if (kb + 64 < SEQ) {
      __builtin_prefetch(Kb + (size_t)(kb + 64 + (lane & 15)) * HDIM, 0, 1);
      __builtin_prefetch(Vb + (size_t)(lane & 15) * SEQ + kb + 64, 0, 1);
    }

    // ---- scores: four 16x16 tiles covering kv [kb, kb+64) ----
    v8f s[4] = {};
#pragma unroll
    for (int t = 0; t < 4; ++t) {
      s[t] = wmma32(aq0, load_frag(Kb + (size_t)(kb + t * 16) * HDIM,      HDIM), s[t]);
      s[t] = wmma32(aq1, load_frag(Kb + (size_t)(kb + t * 16) * HDIM + 32, HDIM), s[t]);
    }

    // ---- online softmax (rows live across the 16 lanes of each half) ----
    float fac[8];
#pragma unroll
    for (int r = 0; r < 8; ++r) {
      float e0 = s[0][r] * sc, e1 = s[1][r] * sc;
      float e2 = s[2][r] * sc, e3 = s[3][r] * sc;
      float mx = fmaxf(fmaxf(e0, e1), fmaxf(e2, e3));
      mx = fmaxf(mx, __shfl_xor(mx, 1, 32));
      mx = fmaxf(mx, __shfl_xor(mx, 2, 32));
      mx = fmaxf(mx, __shfl_xor(mx, 4, 32));
      mx = fmaxf(mx, __shfl_xor(mx, 8, 32));
      const float mn = fmaxf(mrow[r], mx);
      const float p0 = __expf(e0 - mn);
      const float p1 = __expf(e1 - mn);
      const float p2 = __expf(e2 - mn);
      const float p3 = __expf(e3 - mn);
      float rs = (p0 + p1) + (p2 + p3);
      rs += __shfl_xor(rs, 1, 32);
      rs += __shfl_xor(rs, 2, 32);
      rs += __shfl_xor(rs, 4, 32);
      rs += __shfl_xor(rs, 8, 32);
      const float f = __expf(mrow[r] - mn);
      lrow[r] = lrow[r] * f + rs;
      mrow[r] = mn;
      fac[r] = f;
      const int pr = r + half * 8;
      ldsP[w][pr][nlo]      = (_Float16)p0;
      ldsP[w][pr][nlo + 16] = (_Float16)p1;
      ldsP[w][pr][nlo + 32] = (_Float16)p2;
      ldsP[w][pr][nlo + 48] = (_Float16)p3;
    }
#pragma unroll
    for (int j = 0; j < 4; ++j)
#pragma unroll
      for (int r = 0; r < 8; ++r) ctx[j][r] *= fac[r];

    // ensure P staging stores landed before the cross-lane A-fragment reads
    asm volatile("s_wait_dscnt 0" ::: "memory");
    const v16h pa0 = load_frag(&ldsP[w][0][0],  64);   // P: 16(q) x kv[0..31]
    const v16h pa1 = load_frag(&ldsP[w][0][32], 64);   // P: 16(q) x kv[32..63]

    // ---- ctx += P @ V : four 16-wide d chunks, two kv halves each ----
#pragma unroll
    for (int j = 0; j < 4; ++j) {
      v16h vb0 = load_frag(Vb + (size_t)(j * 16) * SEQ + kb,      SEQ);
      v16h vb1 = load_frag(Vb + (size_t)(j * 16) * SEQ + kb + 32, SEQ);
      ctx[j] = wmma32(pa0, vb0, ctx[j]);
      ctx[j] = wmma32(pa1, vb1, ctx[j]);
    }
  }

  // ---- normalize and write ctx tile as f16 [B,S,E] ----
  float inv[8];
#pragma unroll
  for (int r = 0; r < 8; ++r) inv[r] = 1.0f / lrow[r];
  const int b = bh >> 4, h = bh & 15;
#pragma unroll
  for (int j = 0; j < 4; ++j) {
#pragma unroll
    for (int r = 0; r < 8; ++r) {
      const int s = qt * 16 + r + half * 8;
      const int e = h * HDIM + j * 16 + nlo;
      Ctx[((size_t)b * SEQ + s) * EMBED + e] = (_Float16)(ctx[j][r] * inv[r]);
    }
  }
}

// ---------------------------------------------------------------------------
// Host-side orchestration
// Workspace (f16 elems): [0,8M) Xh (reused as Ctx), [8M..12M) weights,
// [12M..20M) Qh, [20M..28M) Kh, [28M..36M) Vth  => 72 MB total.
// ---------------------------------------------------------------------------
extern "C" void kernel_launch(void* const* d_in, const int* in_sizes, int n_in,
                              void* d_out, int out_size, void* d_ws, size_t ws_size,
                              hipStream_t stream) {
  const float* X  = (const float*)d_in[0];
  const float* Wq = (const float*)d_in[1];
  const float* Wk = (const float*)d_in[2];
  const float* Wv = (const float*)d_in[3];
  const float* Wo = (const float*)d_in[4];
  const float* bo = (const float*)d_in[5];
  float* out = (float*)d_out;

  _Float16* ws = (_Float16*)d_ws;
  const size_t M1 = (size_t)1024 * 1024;
  _Float16* Xh  = ws;             // 8M elems; later reused as Ctx (f16)
  _Float16* Wqt = ws + 8 * M1;    // W_q^T  [f,e]
  _Float16* Wkt = ws + 9 * M1;
  _Float16* Wvt = ws + 10 * M1;
  _Float16* Woh = ws + 11 * M1;   // W_o    [f,e] (already B^T layout for x@Wo^T)
  _Float16* Qh  = ws + 12 * M1;   // [B,H,S,D]
  _Float16* Kh  = ws + 20 * M1;   // [B,H,S,D]
  _Float16* Vth = ws + 28 * M1;   // [B,H,D,S]

  // 1) precision conversion / weight re-layout
  cvt_f32_f16  <<<8192, 256, 0, stream>>>(X,  Xh,  MTOT * EMBED);
  transpose_cvt<<<4096, 256, 0, stream>>>(Wq, Wqt, EMBED, EMBED);
  transpose_cvt<<<4096, 256, 0, stream>>>(Wk, Wkt, EMBED, EMBED);
  transpose_cvt<<<4096, 256, 0, stream>>>(Wv, Wvt, EMBED, EMBED);
  cvt_f32_f16  <<<1024, 256, 0, stream>>>(Wo, Woh, EMBED * EMBED);

  // 2) Q/K/V projections (8192x1024x1024 each), 2048 waves / 256 blocks
  gemm_wmma<<<256, 256, 0, stream>>>(Xh, Wqt, Qh,  nullptr, nullptr, MTOT, EMBED, EMBED, 0);
  gemm_wmma<<<256, 256, 0, stream>>>(Xh, Wkt, Kh,  nullptr, nullptr, MTOT, EMBED, EMBED, 0);
  gemm_wmma<<<256, 256, 0, stream>>>(Xh, Wvt, Vth, nullptr, nullptr, MTOT, EMBED, EMBED, 1);

  // 3) flash attention: 8192 q-tiles -> 1024 blocks of 8 waves; Ctx -> Xh
  attn_wmma<<<1024, 256, 0, stream>>>(Qh, Kh, Vth, Xh);

  // 4) output projection + bias, f32 result
  gemm_wmma<<<256, 256, 0, stream>>>(Xh, Woh, nullptr, out, bo, MTOT, EMBED, EMBED, 2);
}
